// TextDecoder_11020886082262
// MI455X (gfx1250) — matrix operand order repaired and checked by reading, not compiled, not run on hardware
//
#include <hip/hip_runtime.h>

typedef unsigned short u16;
typedef unsigned long long u64;
typedef __attribute__((ext_vector_type(16))) __bf16 v16bf;
typedef __attribute__((ext_vector_type(8)))  __bf16 v8bf;
typedef __attribute__((ext_vector_type(8)))  float  v8f;

constexpr int V = 10000, H = 1024, B = 64, T = 32;
constexpr int LROW = 1032;                 // LDS row stride in elems (1024 + 8 pad: 516 dwords, 516%64=4 -> conflict-free)
constexpr int LDS_W_BYTES = 64 * LROW * 2; // 132096 B staged weight tile (64 rows)

__device__ __forceinline__ u16 f2bf(float f) {
  unsigned int u = __float_as_uint(f);
  u += 0x7FFFu + ((u >> 16) & 1u);         // round-to-nearest-even
  return (u16)(u >> 16);
}

__device__ __forceinline__ float sigm(float x) { return 1.0f / (1.0f + __expf(-x)); }

// monotonic uint encoding of float (order-preserving)
__device__ __forceinline__ unsigned mono(float x) {
  unsigned u = __float_as_uint(x);
  return (u & 0x80000000u) ? ~u : (u | 0x80000000u);
}

// ---- CDNA5 async global->LDS copy (16B per lane) + wait ----
__device__ __forceinline__ void async_copy16(void* lds_ptr, const void* gptr) {
  asm volatile("global_load_async_to_lds_b128 %0, %1, off"
               :: "v"((unsigned)(uintptr_t)lds_ptr),
                  "v"((unsigned long long)(uintptr_t)gptr)
               : "memory");
}
__device__ __forceinline__ void async_wait0() {
  asm volatile("s_wait_asynccnt 0" ::: "memory");
}

// A fragment (16x32 bf16, MxK) from global: lane holds row m0+(lane&15);
// K chunks [k0+8*half,+8) and [k0+16+8*half,+8).
__device__ __forceinline__ v16bf load_a(const u16* __restrict__ base, int row, int k0, int lane) {
  const int half = lane >> 4;
  const u16* p = base + (size_t)row * H + k0 + half * 8;
  union { v16bf v; v8bf h[2]; } u;
  u.h[0] = *(const v8bf*)(p);
  u.h[1] = *(const v8bf*)(p + 16);
  return u.v;
}

// B fragment (32x16 bf16, KxN) from the staged LDS tile: lane holds W row (lrow base + lane&15),
// K chunks [k0+8*half,+8) and [k0+16+8*half,+8).
__device__ __forceinline__ v16bf load_b_lds(const u16* s, int lrow, int k0, int half) {
  const u16* p = s + lrow * LROW + k0 + half * 8;
  union { v16bf v; v8bf h[2]; } u;
  u.h[0] = *(const v8bf*)(p);
  u.h[1] = *(const v8bf*)(p + 16);
  return u.v;
}

// ---------------- weight conversion fp32 -> bf16 ----------------
__global__ void cvt_kernel(const float* __restrict__ src, u16* __restrict__ dst, int n) {
  int i = blockIdx.x * blockDim.x + threadIdx.x;
  if (i < n) dst[i] = f2bf(src[i]);
}

// ---------------- state init ----------------
__global__ void init_kernel(const float* __restrict__ hidden, const float* __restrict__ cell,
                            u16* __restrict__ h1bf, u16* __restrict__ h2bf,
                            float* __restrict__ c1, float* __restrict__ c2,
                            u64* __restrict__ amax) {
  int i = blockIdx.x * blockDim.x + threadIdx.x;
  if (i < B * H) {
    h1bf[i] = f2bf(hidden[i]);
    h2bf[i] = f2bf(hidden[B * H + i]);
    c1[i]   = cell[i];
    c2[i]   = cell[B * H + i];
  }
  if (i < T * B) amax[i] = 0ull;
}

// ---------------- out[:,0,:] = one_hot(captions[:,0]) ----------------
__global__ void out0_kernel(const int* __restrict__ captions, float* __restrict__ out) {
  int i = blockIdx.x * blockDim.x + threadIdx.x;
  if (i < B * V) {
    int b = i / V, v = i % V;
    out[(size_t)b * T * V + v] = (v == captions[b * T]) ? 1.0f : 0.0f;
  }
}

// ---------------- fused LSTM layer ----------------
// gates[b, g*H+n] = A0@W0^T (+ A1@W1^T) (+ one-hot gather, token selected from
// teacher-forcing mask or previous step's atomic argmax) + bih + bhh, then cell update.
// grid = H/16 blocks (N tiles), 128 threads = 4 waves (M tiles of 16, B=64).
// Weight tiles (4 gates x 16 rows x K=1024) staged in LDS via async DMA, shared by all 4 waves.
__global__ __launch_bounds__(128) void lstm_kernel(
    const u16* __restrict__ A0, const u16* __restrict__ W0,
    const u16* __restrict__ A1, const u16* __restrict__ W1,
    const float* __restrict__ Woh, const int* __restrict__ caps,
    const unsigned char* __restrict__ tf, const u64* __restrict__ amax, int t,
    const float* __restrict__ bih, const float* __restrict__ bhh,
    float* __restrict__ c, u16* __restrict__ hout) {
  extern __shared__ u16 sW[];                 // 64 rows x LROW elems
  const int tid  = threadIdx.x;
  const int lane = tid & 31;
  const int m0   = (tid >> 5) << 4;
  const int n0   = blockIdx.x << 4;
  const int nr   = lane & 15;
  const int half = lane >> 4;

  v8f acc[4] = {};

  auto stage = [&](const u16* __restrict__ Wsrc) {
    for (int i = 0; i < 64; ++i) {            // 8192 x 16B = 128KB per block
      int slot = i * 128 + tid;
      int row = slot >> 7, w = slot & 127;    // 128 x 16B slots per row
      int grow = (row >> 4) * H + n0 + (row & 15);
      async_copy16((char*)sW + row * (LROW * 2) + w * 16,
                   Wsrc + (size_t)grow * H + (size_t)w * 8);
    }
    async_wait0();
    __syncthreads();
  };
  auto gemm = [&](const u16* __restrict__ A) {
    for (int k0 = 0; k0 < H; k0 += 32) {
      v16bf a = load_a(A, m0 + nr, k0, lane);
#pragma unroll
      for (int g = 0; g < 4; ++g) {
        v16bf b = load_b_lds(sW, g * 16 + nr, k0, half);
        acc[g] = __builtin_amdgcn_wmma_f32_16x16x32_bf16(false, a, false, b,
                                                         (short)0, acc[g], false, false);
      }
    }
  };

  stage(W0);
  gemm(A0);
  if (A1) {
    __syncthreads();                          // all waves done reading W0 tile
    stage(W1);
    gemm(A1);
  }

  const int nglob = n0 + nr;
  float bias[4];
#pragma unroll
  for (int g = 0; g < 4; ++g) bias[g] = bih[g * H + nglob] + bhh[g * H + nglob];

#pragma unroll
  for (int r = 0; r < 8; ++r) {
    const int m = m0 + r + 8 * half;
    float gi = acc[0][r] + bias[0];
    float gf = acc[1][r] + bias[1];
    float gg = acc[2][r] + bias[2];
    float go = acc[3][r] + bias[3];
    if (Woh) {  // one-hot input: select token, gather column of W_ih_l0 [4H, V]
      int tok;
      if (t == 1 || tf[t - 1]) tok = caps[m * T + (t - 1)];
      else                     tok = (int)(~(unsigned)amax[(size_t)(t - 1) * B + m]);
      gi += Woh[(size_t)(0 * H + nglob) * V + tok];
      gf += Woh[(size_t)(1 * H + nglob) * V + tok];
      gg += Woh[(size_t)(2 * H + nglob) * V + tok];
      go += Woh[(size_t)(3 * H + nglob) * V + tok];
    }
    const size_t idx = (size_t)m * H + nglob;
    float cn = sigm(gf) * c[idx] + sigm(gi) * tanhf(gg);
    c[idx] = cn;
    hout[idx] = f2bf(sigm(go) * tanhf(cn));
  }
}

// ---------------- logits = h2 @ W_out^T + b_out -> out[:, t, :], fused argmax ----------------
// Each wave computes 4 N-tiles for its M-tile (A frag reused 4x); the block's 64 W rows are
// staged in LDS once; per-row block max is pushed to amax[t*B+row] via atomic max of
// (mono(value)<<32 | ~col) -> exact argmax with first-index tie-break.
__global__ __launch_bounds__(128) void logits_kernel(
    const u16* __restrict__ A, const u16* __restrict__ Wb,
    const float* __restrict__ bias, float* __restrict__ out,
    u64* __restrict__ amax, int t) {
  extern __shared__ u16 smem[];
  u16* sW = smem;
  u64* keys = (u64*)((char*)smem + LDS_W_BYTES);   // [64 rows][64 cols]
  const int tid  = threadIdx.x;
  const int lane = tid & 31;
  const int m0   = (tid >> 5) << 4;
  const int nr   = lane & 15;
  const int half = lane >> 4;
  const int ntb  = (blockIdx.x * 4 < 621) ? blockIdx.x * 4 : 621;  // tail clamp (dup work is idempotent)
  const int row0 = ntb * 16;

  for (int i = 0; i < 64; ++i) {               // stage 64 contiguous W_out rows (128KB)
    int slot = i * 128 + tid;
    int row = slot >> 7, w = slot & 127;
    async_copy16((char*)sW + row * (LROW * 2) + w * 16,
                 Wb + (size_t)(row0 + row) * H + (size_t)w * 8);
  }
  async_wait0();
  __syncthreads();

  v8f acc[4] = {};
  for (int k0 = 0; k0 < H; k0 += 32) {
    v16bf a = load_a(A, m0 + nr, k0, lane);
#pragma unroll
    for (int j = 0; j < 4; ++j) {
      v16bf b = load_b_lds(sW, j * 16 + nr, k0, half);
      acc[j] = __builtin_amdgcn_wmma_f32_16x16x32_bf16(false, a, false, b,
                                                       (short)0, acc[j], false, false);
    }
  }

#pragma unroll
  for (int j = 0; j < 4; ++j) {
    const int nglob = row0 + j * 16 + nr;
    const float bv = bias[nglob];
#pragma unroll
    for (int r = 0; r < 8; ++r) {
      const int m = m0 + r + 8 * half;
      float x = acc[j][r] + bv;
      out[(size_t)m * T * V + (size_t)t * V + nglob] = x;
      keys[m * 64 + j * 16 + nr] = ((u64)mono(x) << 32) | (unsigned)(~(unsigned)nglob);
    }
  }
  __syncthreads();
  if (tid < 64) {                              // per-row block max -> global atomic max
    u64 best = 0;
    const u64* krow = keys + tid * 64;
#pragma unroll 4
    for (int j = 0; j < 64; ++j) { u64 k = krow[j]; best = (k > best) ? k : best; }
    atomicMax(&amax[(size_t)t * B + tid], best);
  }
}

extern "C" void kernel_launch(void* const* d_in, const int* in_sizes, int n_in,
                              void* d_out, int out_size, void* d_ws, size_t ws_size,
                              hipStream_t stream) {
  (void)in_sizes; (void)n_in; (void)out_size; (void)ws_size;
  const float* hidden  = (const float*)d_in[0];
  const float* cellin  = (const float*)d_in[1];
  const int*   caps    = (const int*)d_in[2];
  const unsigned char* tf = (const unsigned char*)d_in[3];
  const float* W_ih_l0 = (const float*)d_in[4];
  const float* W_hh_l0 = (const float*)d_in[5];
  const float* b_ih_l0 = (const float*)d_in[6];
  const float* b_hh_l0 = (const float*)d_in[7];
  const float* W_ih_l1 = (const float*)d_in[8];
  const float* W_hh_l1 = (const float*)d_in[9];
  const float* b_ih_l1 = (const float*)d_in[10];
  const float* b_hh_l1 = (const float*)d_in[11];
  const float* W_out   = (const float*)d_in[12];
  const float* b_out   = (const float*)d_in[13];
  float* out = (float*)d_out;

  // workspace carve-out (256B aligned), ~45 MB total, L2-resident
  char* p = (char*)d_ws;
  auto carve = [&](size_t bytes) { char* r = p; p += (bytes + 255) & ~(size_t)255; return r; };
  u16* Whh0b = (u16*)carve((size_t)4 * H * H * 2);
  u16* Wih1b = (u16*)carve((size_t)4 * H * H * 2);
  u16* Whh1b = (u16*)carve((size_t)4 * H * H * 2);
  u16* Woutb = (u16*)carve((size_t)V * H * 2);
  u16* h1bf[2] = { (u16*)carve(B * H * 2), (u16*)carve(B * H * 2) };
  u16* h2bf[2] = { (u16*)carve(B * H * 2), (u16*)carve(B * H * 2) };
  float* c1 = (float*)carve(B * H * 4);
  float* c2 = (float*)carve(B * H * 4);
  u64* amax = (u64*)carve((size_t)T * B * 8);

  int n = 4 * H * H;
  cvt_kernel<<<(n + 255) / 256, 256, 0, stream>>>(W_hh_l0, Whh0b, n);
  cvt_kernel<<<(n + 255) / 256, 256, 0, stream>>>(W_ih_l1, Wih1b, n);
  cvt_kernel<<<(n + 255) / 256, 256, 0, stream>>>(W_hh_l1, Whh1b, n);
  n = V * H;
  cvt_kernel<<<(n + 255) / 256, 256, 0, stream>>>(W_out, Woutb, n);

  init_kernel<<<(B * H + 255) / 256, 256, 0, stream>>>(hidden, cellin,
                                                       h1bf[0], h2bf[0], c1, c2, amax);
  out0_kernel<<<(B * V + 255) / 256, 256, 0, stream>>>(caps, out);

  int cur = 0;
  for (int t = 1; t < T; ++t) {
    const int nb = cur ^ 1;
    // layer 0: token select + one-hot gather + h1 @ W_hh_l0^T
    lstm_kernel<<<H / 16, 128, LDS_W_BYTES, stream>>>(
        h1bf[cur], Whh0b, nullptr, nullptr, W_ih_l0, caps, tf, amax, t,
        b_ih_l0, b_hh_l0, c1, h1bf[nb]);
    // layer 1: h1_new @ W_ih_l1^T + h2 @ W_hh_l1^T
    lstm_kernel<<<H / 16, 128, LDS_W_BYTES, stream>>>(
        h1bf[nb], Wih1b, h2bf[cur], Whh1b, nullptr, caps, tf, amax, t,
        b_ih_l1, b_hh_l1, c2, h2bf[nb]);
    // logits -> out[:, t, :] with fused atomic argmax
    logits_kernel<<<157, 128, LDS_W_BYTES + 64 * 64 * 8, stream>>>(
        h2bf[nb], Woutb, b_out, out, amax, t);
    cur = nb;
  }
}
